// PiOModel_317827580649
// MI455X (gfx1250) — matrix operand order repaired and checked by reading, not compiled, not run on hardware
//
#include <hip/hip_runtime.h>
#include <hip/hip_bf16.h>

typedef __attribute__((ext_vector_type(16))) _Float16 v16h;
typedef __attribute__((ext_vector_type(8)))  _Float16 v8h;
typedef __attribute__((ext_vector_type(8)))  float    v8f;
typedef __attribute__((ext_vector_type(4)))  unsigned u32x4;
typedef __attribute__((ext_vector_type(8)))  unsigned u32x8;

constexpr int NB = 2, NN = 256, NM = 32, ND = 64, NL = 4, NOUT = 10;

// ---------------------------------------------------------------- WMMA helpers

__device__ __forceinline__ v8f wmma_f16f32(v16h a, v16h b, v8f c) {
  return __builtin_amdgcn_wmma_f32_16x16x32_f16(false, a, false, b, (short)0, c,
                                                false, false);
}

// A fragment, 16x32 f16, row-major source (row stride in halves).
// lane<16: row=lane, elems 0..7 = K0..7, 8..15 = K16..23
// lane>=16: row=lane-16, elems 0..7 = K8..15, 8..15 = K24..31
__device__ __forceinline__ v16h frag_ld_a(const _Float16* base, int rstride) {
  int lane = threadIdx.x & 31;
  const _Float16* p = base + (lane & 15) * rstride + ((lane >> 4) << 3);
  v8h lo = *(const v8h*)p;
  v8h hi = *(const v8h*)(p + 16);
  v16h o;
#pragma unroll
  for (int e = 0; e < 8; ++e) { o[e] = lo[e]; o[e + 8] = hi[e]; }
  return o;
}

// B fragment, 32x16 f16, column-major source (col stride in halves; K contiguous).
// lane<16: col=lane, elems 0..7 = K0..7, 8..15 = K16..23 ; lane>=16: col=lane-16, elems 0..7 = K8..15, 8..15 = K24..31
__device__ __forceinline__ v16h frag_ld_b(const _Float16* base, int cstride) {
  int lane = threadIdx.x & 31;
  const _Float16* p = base + (lane & 15) * cstride + ((lane >> 4) << 3);
  v8h lo = *(const v8h*)p;
  v8h hi = *(const v8h*)(p + 16);
  v16h o;
#pragma unroll
  for (int e = 0; e < 8; ++e) { o[e] = lo[e]; o[e + 8] = hi[e]; }
  return o;
}

// ---------------------------------------------------- TDM: 64x64 f16 tile -> LDS
// D# per CDNA5 ISA. 2D tensor, data_size=2B,
// tile_dim0=64, tile_dim1=64, tensor_dim0_stride=64. Group2/3 NULL (2D).
__device__ __forceinline__ void tdm_load_w64(const _Float16* gsrc,
                                             unsigned lds_byte_off) {
  unsigned long long ga = (unsigned long long)(uintptr_t)gsrc;
  u32x4 g0;
  g0[0] = 1u;                                  // count=1, user mode
  g0[1] = lds_byte_off;                        // lds_addr
  g0[2] = (unsigned)(ga & 0xffffffffu);        // global_addr[31:0]
  g0[3] = (unsigned)((ga >> 32) & 0x01ffffffu) // global_addr[56:32]
          | (2u << 30);                        // type=2 (image)
  u32x8 g1;
  g1[0] = (1u << 16);          // workgroup_mask=0, data_size=1 (2 bytes)
  g1[1] = (64u << 16);         // tensor_dim0 = 64 (bits 79:48 low half)
  g1[2] = (64u << 16);         // tensor_dim0 hi=0 | tensor_dim1 = 64 (low half)
  g1[3] = (64u << 16);         // tensor_dim1 hi=0 | tile_dim0 = 64
  g1[4] = 64u;                 // tile_dim1 = 64, tile_dim2 = 0
  g1[5] = 64u;                 // tensor_dim0_stride = 64 (low 32)
  g1[6] = 0u;                  // stride0 hi | tensor_dim1_stride lo
  g1[7] = 0u;
  asm volatile("tensor_load_to_lds %0, %1" ::"s"(g0), "s"(g1) : "memory");
}

// ------------------------------------------------- generic rows x 64 @ 64 x 64
// A: f16 rows x 64 row-major. WT: f16 [n][k] (weight pre-transposed, n-major).
// mode 0: outF f32 row-major; mode 1: outH f16 row-major;
// mode 2: outH f16 transposed batched layout  outH[(batch*64+col)*rpb + row%rpb]
__global__ void k_gemm64(const _Float16* __restrict__ A,
                         const _Float16* __restrict__ WT,
                         const float* __restrict__ bias,
                         float* __restrict__ outF, _Float16* __restrict__ outH,
                         int mode, int relu, int rpb) {
  const int wave = threadIdx.x >> 5;
  const int lane = threadIdx.x & 31;
  const int rowBase = blockIdx.x * 32 + (wave & 1) * 16;
  const int ct = wave >> 1;  // 4 column tiles of 16
  const _Float16* Ap = A + (size_t)rowBase * 64;
  __builtin_prefetch(Ap + 32 * 64, 0, 1);  // next block's A rows -> global_prefetch
  v8f acc = {};
#pragma unroll
  for (int ks = 0; ks < 2; ++ks) {
    v16h a = frag_ld_a(Ap + ks * 32, 64);
    v16h b = frag_ld_b(WT + ct * 16 * 64 + ks * 32, 64);
    acc = wmma_f16f32(a, b, acc);
  }
  const int col = ct * 16 + (lane & 15);
  const float bv = bias ? bias[col] : 0.0f;
  const int rhi = (lane >> 4) << 3;
#pragma unroll
  for (int r = 0; r < 8; ++r) {
    float x = acc[r] + bv;
    if (relu) x = fmaxf(x, 0.0f);
    acc[r] = x;
  }
  if (mode == 0) {
#pragma unroll
    for (int r = 0; r < 8; ++r)
      outF[(size_t)(rowBase + rhi + r) * 64 + col] = acc[r];
  } else if (mode == 1) {
#pragma unroll
    for (int r = 0; r < 8; ++r)
      outH[(size_t)(rowBase + rhi + r) * 64 + col] = (_Float16)acc[r];
  } else {
    const int bb = rowBase / rpb;
    const int rin = rowBase - bb * rpb + rhi;
    v8h pk;
#pragma unroll
    for (int r = 0; r < 8; ++r) pk[r] = (_Float16)acc[r];
    *(v8h*)(outH + (size_t)(bb * 64 + col) * rpb + rin) = pk;
  }
}

// ------------------------------------------------------------- fused edge kernel
// Per 16x16 (i,j) tile: vv via 64 WMMA-K32 (one per d), e = s1*s2*vv,
// el(+)= mlp(e)*gsn. el f32 (b,i,j,d) + elt f16 (b,d,i,j).
// MLP weights staged into LDS by the Tensor Data Mover (wave 0, async).
__global__ void k_el(const _Float16* __restrict__ v1t,
                     const _Float16* __restrict__ v2t,
                     const float* __restrict__ s1, const float* __restrict__ s2,
                     const _Float16* __restrict__ W1t, const float* __restrict__ b1,
                     const _Float16* __restrict__ W2t, const float* __restrict__ b2,
                     const float* __restrict__ gsn, float* __restrict__ el,
                     _Float16* __restrict__ elt, int first) {
  extern __shared__ char smemRaw[];
  _Float16* eL = (_Float16*)smemRaw;                   // [0, 36864)  256 x 72
  _Float16* hL = eL + 256 * 72;                        // [36864, 73728)
  float* s1L = (float*)(smemRaw + 73728);              // 16 x 64
  float* s2L = s1L + 16 * 64;                          // up to 81920
  _Float16* w1L = (_Float16*)(smemRaw + 81920);        // 64 x 64 f16 (8KB)
  _Float16* w2L = (_Float16*)(smemRaw + 90112);        // 64 x 64 f16 (8KB)

  const int b = blockIdx.z, i0 = blockIdx.y * 16, j0 = blockIdx.x * 16;
  const int wave = threadIdx.x >> 5, lane = threadIdx.x & 31;

  // Kick off async TDM weight staging (one wave issues; EXEC-independent DMA).
  if (wave == 0) {
    tdm_load_w64(W1t, 81920u);
    tdm_load_w64(W2t, 90112u);
  }

  for (int t = threadIdx.x; t < 1024; t += 256) {
    int r = t >> 6, c = t & 63;
    s1L[t] = s1[((size_t)b * NN + i0 + r) * ND + c];
    s2L[t] = s2[((size_t)b * NN + j0 + r) * ND + c];
  }
  __syncthreads();

  // Phase 1: vv (K = M = 32) for 8 d per wave, scaled into eL (f16)
  const int jl = lane & 15, rhi = (lane >> 4) << 3;
#pragma unroll
  for (int dd = 0; dd < 8; ++dd) {
    int d = wave * 8 + dd;
    const _Float16* ap = v1t + (((size_t)b * 64 + d) * NN + i0) * NM;
    const _Float16* bp = v2t + (((size_t)b * 64 + d) * NN + j0) * NM;
    v8f c = {};
    c = wmma_f16f32(frag_ld_a(ap, NM), frag_ld_b(bp, NM), c);
#pragma unroll
    for (int r = 0; r < 8; ++r) {
      int il = r + rhi;
      float e = s1L[il * 64 + d] * s2L[jl * 64 + d] * c[r];
      eL[(il * 16 + jl) * 72 + d] = (_Float16)e;
    }
  }
  // TDM completion must precede phase 2 weight reads (per-wave counter on
  // the issuing wave; barrier publishes the LDS writes to all waves).
  if (wave == 0) __builtin_amdgcn_s_wait_tensorcnt(0);
  __syncthreads();

  // Phase 2: h = relu(e @ W1 + b1), rows = 256 pair-rows, cols = 64
  const int ct = wave >> 1;
  const int rt0 = (wave & 1) * 8;
  for (int tt = 0; tt < 8; ++tt) {
    int rt = rt0 + tt;
    v8f c = {};
#pragma unroll
    for (int ks = 0; ks < 2; ++ks) {
      v16h a = frag_ld_a(&eL[(rt * 16) * 72 + ks * 32], 72);
      v16h w = frag_ld_b(w1L + ct * 16 * 64 + ks * 32, 64);
      c = wmma_f16f32(a, w, c);
    }
    int col = ct * 16 + (lane & 15);
    float bv = b1[col];
#pragma unroll
    for (int r = 0; r < 8; ++r) {
      float h = fmaxf(c[r] + bv, 0.0f);
      hL[(rt * 16 + rhi + r) * 72 + col] = (_Float16)h;
    }
  }
  __syncthreads();

  // Phase 3: out = h @ W2 + b2, scale by gsn, accumulate el; write elt (f16)
  const float g = gsn[b];
  for (int tt = 0; tt < 8; ++tt) {
    int rt = rt0 + tt;  // = il for this row tile
    v8f c = {};
#pragma unroll
    for (int ks = 0; ks < 2; ++ks) {
      v16h a = frag_ld_a(&hL[(rt * 16) * 72 + ks * 32], 72);
      v16h w = frag_ld_b(w2L + ct * 16 * 64 + ks * 32, 64);
      c = wmma_f16f32(a, w, c);
    }
    int col = ct * 16 + (lane & 15);  // = d
    float bv = b2[col];
    int il = rt;
    size_t elBase = (((size_t)b * NN + i0 + il) * NN + j0) * ND + col;
    size_t eltBase = (((size_t)b * 64 + col) * NN + (i0 + il)) * NN + j0 + rhi;
    v8h pk;
#pragma unroll
    for (int r = 0; r < 8; ++r) {
      int jj = rhi + r;
      float val = (c[r] + bv) * g;
      size_t ei = elBase + (size_t)jj * ND;
      float nv = first ? val : (el[ei] + val);
      el[ei] = nv;
      pk[r] = (_Float16)nv;
    }
    *(v8h*)(elt + eltBase) = pk;
  }
}

// ------------------------------------------------ vo = einsum('bijd,bjmd->bimd')
// Per (b,d): (N x N) @ (N x M). elt f16 (b,d,i,j); wt f16 (b,d,m,j) so the
// B operand is K-contiguous (vector loads, no scalar gather).
__global__ void k_vo(const _Float16* __restrict__ elt,
                     const _Float16* __restrict__ wt, float* __restrict__ coord) {
  const int bd = blockIdx.x;
  const int b = bd >> 6, d = bd & 63;
  const int wave = threadIdx.x >> 5, lane = threadIdx.x & 31;
  const int mt = wave & 1;  // 2 m-tiles of 16
  const _Float16* eb = elt + (size_t)bd * NN * NN;
  const _Float16* wb = wt + (size_t)bd * NN * NM;
  for (int tt = 0; tt < 4; ++tt) {
    int it = (wave >> 1) + tt * 4;  // 16 i-tiles
    __builtin_prefetch(eb + (size_t)(it * 16) * NN + 4096, 0, 1);
    v8f acc = {};
#pragma unroll
    for (int ks = 0; ks < 8; ++ks) {
      v16h a = frag_ld_a(eb + (size_t)(it * 16) * NN + ks * 32, NN);
      v16h w = frag_ld_b(wb + (size_t)(mt * 16) * NN + ks * 32, NN);
      acc = wmma_f16f32(a, w, acc);
    }
    int m = mt * 16 + (lane & 15);
    int rhi = (lane >> 4) << 3;
#pragma unroll
    for (int r = 0; r < 8; ++r) {
      int n = it * 16 + rhi + r;
      size_t ci = (((size_t)b * NN + n) * NM + m) * ND + d;
      coord[ci] += acc[r];
    }
  }
}

// ---------------------------------------------------------------- small kernels

__global__ void k_gsn(const unsigned char* __restrict__ nodemask,
                      float* __restrict__ gsn) {
  __shared__ int cnt;
  int b = blockIdx.x;
  if (threadIdx.x == 0) cnt = 0;
  __syncthreads();
  atomicAdd(&cnt, nodemask[b * NN + threadIdx.x] ? 1 : 0);
  __syncthreads();
  if (threadIdx.x == 0) gsn[b] = rsqrtf((float)(NN - cnt));
}

__global__ void k_lambda(const float* __restrict__ LE,
                         const float* __restrict__ Wphi,
                         const float* __restrict__ bphi,
                         const float* __restrict__ Wrho,
                         const float* __restrict__ brho,
                         const unsigned char* __restrict__ lmask,
                         float* __restrict__ le) {
  __shared__ float h[NM * ND];
  __shared__ float pooled[ND];
  int b = blockIdx.x;
  const float* Lb = LE + (size_t)b * NM * ND;
  for (int t = threadIdx.x; t < NM * ND; t += 256) {
    int m = t >> 6, d = t & 63;
    float a = bphi[d];
    for (int k = 0; k < ND; ++k) a += Lb[m * ND + k] * Wphi[k * ND + d];
    h[t] = fmaxf(a, 0.0f);
  }
  __syncthreads();
  if (threadIdx.x < ND) {
    float s = 0.0f, c = 0.0f;
    for (int m = 0; m < NM; ++m) {
      float w = lmask[b * NM + m] ? 0.0f : 1.0f;
      s += h[m * ND + threadIdx.x] * w;
      c += w;
    }
    pooled[threadIdx.x] = s / fmaxf(c, 1.0f);
  }
  __syncthreads();
  for (int t = threadIdx.x; t < NM * ND; t += 256) {
    int m = t >> 6, d = t & 63;
    float a = brho[d];
    for (int k = 0; k < ND; ++k) a += (h[m * ND + k] + pooled[k]) * Wrho[k * ND + d];
    le[(size_t)b * NM * ND + t] = lmask[b * NM + m] ? 0.0f : a;
  }
}

__global__ void k_coord(const float* __restrict__ U, const float* __restrict__ le,
                        float* __restrict__ coord) {
  __shared__ float us[NM];
  int b = blockIdx.y, n = blockIdx.x;
  if (threadIdx.x < NM) us[threadIdx.x] = U[(((size_t)b * NN) + n) * NM + threadIdx.x];
  __syncthreads();
  size_t base = (((size_t)b * NN) + n) * NM * ND;
  for (int t = threadIdx.x; t < NM * ND; t += 256)
    coord[base + t] = us[t >> 6] * le[(size_t)b * NM * ND + t];
}

__global__ void k_vln(const float* __restrict__ coord, float* __restrict__ v,
                      _Float16* __restrict__ vh) {
  __shared__ float c[NM * ND];
  __shared__ float nrm[ND];
  int b = blockIdx.y, n = blockIdx.x;
  size_t base = (((size_t)b * NN) + n) * NM * ND;
  for (int t = threadIdx.x; t < NM * ND; t += 256) c[t] = coord[base + t];
  __syncthreads();
  if (threadIdx.x < NM) {
    float m = 0.0f;
    for (int d = 0; d < ND; ++d) m += c[threadIdx.x * ND + d];
    m *= (1.0f / ND);
    for (int d = 0; d < ND; ++d) c[threadIdx.x * ND + d] -= m;
  }
  __syncthreads();
  if (threadIdx.x < ND) {
    float s = 0.0f;
    for (int m = 0; m < NM; ++m) {
      float x = c[m * ND + threadIdx.x];
      s += x * x;
    }
    nrm[threadIdx.x] = 1.0f / fmaxf(sqrtf(s), 1e-3f);
  }
  __syncthreads();
  for (int t = threadIdx.x; t < NM * ND; t += 256) {
    float x = c[t] * nrm[t & 63];
    v[base + t] = x;
    vh[base + t] = (_Float16)x;
  }
}

__global__ void k_sln(const float* __restrict__ X, float* __restrict__ sF,
                      _Float16* __restrict__ sH) {
  __shared__ float s[ND];
  __shared__ float mv[2];
  int b = blockIdx.y, n = blockIdx.x, t = threadIdx.x;
  size_t base = (((size_t)b * NN) + n) * ND;
  float x = X[base + t];
  s[t] = x;
  __syncthreads();
  if (t == 0) {
    float m = 0.0f;
    for (int k = 0; k < ND; ++k) m += s[k];
    m *= (1.0f / ND);
    float vv = 0.0f;
    for (int k = 0; k < ND; ++k) { float d = s[k] - m; vv += d * d; }
    vv *= (1.0f / ND);
    mv[0] = m;
    mv[1] = rsqrtf(vv + 1e-5f);
  }
  __syncthreads();
  float o = (x - mv[0]) * mv[1];
  sF[base + t] = o;
  sH[base + t] = (_Float16)o;
}

__global__ void k_vprod(const float* __restrict__ p1, const float* __restrict__ p2,
                        float* __restrict__ vp) {
  int b = blockIdx.y, n = blockIdx.x, d = threadIdx.x;
  size_t base = (((size_t)b * NN) + n) * NM * ND;
  float s = 0.0f;
  for (int m = 0; m < NM; ++m) {
    size_t i = base + m * ND + d;
    s += p1[i] * p2[i];
  }
  vp[(((size_t)b * NN) + n) * ND + d] = s;
}

__global__ void k_ts(const float* __restrict__ sb, const float* __restrict__ m1,
                     const float* __restrict__ vp, _Float16* __restrict__ tsh, int n) {
  int i = blockIdx.x * 256 + threadIdx.x;
  if (i < n) tsh[i] = (_Float16)(sb[i] + m1[i] * vp[i]);
}

// tv = v * (1 + m2) ; emit f16 copy PERMUTED to (b, m, n, d) so that the
// mode-2 GEMM produces wT in (b, d, m, j) layout (K-contiguous for k_vo).
__global__ void k_tv(const float* __restrict__ v, const float* __restrict__ m2,
                     _Float16* __restrict__ tvh, int n) {
  int i = blockIdx.x * 256 + threadIdx.x;
  if (i < n) {
    int d = i & 63;
    int m = (i >> 6) & 31;
    int nn = (i >> 11) & 255;
    int b = i >> 19;
    int bn = i >> 11;  // (b,n)
    float x = v[i] * (1.0f + m2[bn * ND + d]);
    tvh[((((size_t)b * NM + m) * NN) + nn) * ND + d] = (_Float16)x;
  }
}

__global__ void k_so(const float* __restrict__ el, const float* __restrict__ g,
                     float* __restrict__ X) {
  __shared__ float red[256];
  int b = blockIdx.y, i = blockIdx.x;
  int d = threadIdx.x & 63, jq = threadIdx.x >> 6;
  float s = 0.0f;
  for (int j = jq; j < NN; j += 4)
    s += el[((((size_t)b * NN) + i) * NN + j) * ND + d] *
         g[(((size_t)b * NN) + j) * ND + d];
  red[threadIdx.x] = s;
  __syncthreads();
  if (threadIdx.x < ND) {
    float t = red[d] + red[64 + d] + red[128 + d] + red[192 + d];
    X[(((size_t)b * NN) + i) * ND + d] += t;
  }
}

__global__ void k_tohalf(const float* __restrict__ in, _Float16* __restrict__ out,
                         int n) {
  int i = blockIdx.x * 256 + threadIdx.x;
  if (i < n) out[i] = (_Float16)in[i];
}

__global__ void k_copyf(const float* __restrict__ in, float* __restrict__ out,
                        int n) {
  int i = blockIdx.x * 256 + threadIdx.x;
  if (i < n) out[i] = in[i];
}

// transpose DxD f32 weight -> f16 n-major WT[n][k]; one matrix per block
__global__ void k_prepw(const float* __restrict__ W, _Float16* __restrict__ WT) {
  int mat = blockIdx.x;
  const float* w = W + (size_t)mat * ND * ND;
  _Float16* o = WT + (size_t)mat * ND * ND;
  for (int t = threadIdx.x; t < ND * ND; t += 256) {
    int n = t >> 6, k = t & 63;
    o[t] = (_Float16)w[k * ND + n];
  }
}

__global__ void k_pool(const float* __restrict__ X,
                       const unsigned char* __restrict__ nmask,
                       float* __restrict__ Xp) {
  int b = blockIdx.x, d = threadIdx.x;
  float s = 0.0f, c = 0.0f;
  for (int n = 0; n < NN; ++n) {
    float w = nmask[b * NN + n] ? 0.0f : 1.0f;
    s += X[(((size_t)b * NN) + n) * ND + d] * w;
    c += w;
  }
  Xp[b * ND + d] = s / fmaxf(c, 1.0f);
}

__global__ void k_pred(const float* __restrict__ Xp, const float* __restrict__ W1,
                       const float* __restrict__ b1, const float* __restrict__ W2,
                       const float* __restrict__ b2, float* __restrict__ out) {
  __shared__ float h[ND];
  int b = blockIdx.x, t = threadIdx.x;
  float a = b1[t];
  for (int k = 0; k < ND; ++k) a += Xp[b * ND + k] * W1[k * ND + t];
  h[t] = fmaxf(a, 0.0f);
  __syncthreads();
  if (t < NOUT) {
    float o = b2[t];
    for (int k = 0; k < ND; ++k) o += h[k] * W2[k * NOUT + t];
    out[b * NOUT + t] = o;
  }
}

// ------------------------------------------------------------------- host side

extern "C" void kernel_launch(void* const* d_in, const int* in_sizes, int n_in,
                              void* d_out, int out_size, void* d_ws, size_t ws_size,
                              hipStream_t stream) {
  (void)in_sizes; (void)n_in; (void)out_size; (void)ws_size;
  const float* LambdaEmb = (const float*)d_in[0];
  const float* U = (const float*)d_in[1];
  const float* X = (const float*)d_in[2];
  const float* Wphi = (const float*)d_in[3];   const float* bphi = (const float*)d_in[4];
  const float* Wrho = (const float*)d_in[5];   const float* brho = (const float*)d_in[6];
  const float* el_Ws1 = (const float*)d_in[7]; const float* el_bs1 = (const float*)d_in[8];
  const float* el_Ws2 = (const float*)d_in[9]; const float* el_bs2 = (const float*)d_in[10];
  const float* el_Wv1 = (const float*)d_in[11];const float* el_Wv2 = (const float*)d_in[12];
  const float* el_mW1 = (const float*)d_in[13];const float* el_mb1 = (const float*)d_in[14];
  const float* el_mW2 = (const float*)d_in[15];const float* el_mb2 = (const float*)d_in[16];
  const float* mx_Wv1 = (const float*)d_in[17];const float* mx_Wv2 = (const float*)d_in[18];
  const float* mx_s1W1 = (const float*)d_in[19];const float* mx_s1b1 = (const float*)d_in[20];
  const float* mx_s1W2 = (const float*)d_in[21];const float* mx_s1b2 = (const float*)d_in[22];
  const float* mx_s2W1 = (const float*)d_in[23];const float* mx_s2b1 = (const float*)d_in[24];
  const float* mx_s2W2 = (const float*)d_in[25];const float* mx_s2b2 = (const float*)d_in[26];
  const float* cv_W1 = (const float*)d_in[27]; const float* cv_b1 = (const float*)d_in[28];
  const float* cv_W2 = (const float*)d_in[29]; const float* cv_b2 = (const float*)d_in[30];
  const float* cv_Wv = (const float*)d_in[31];
  const float* pred_W1 = (const float*)d_in[32];const float* pred_b1 = (const float*)d_in[33];
  const float* pred_W2 = (const float*)d_in[34];const float* pred_b2 = (const float*)d_in[35];
  const unsigned char* LambdaMask = (const unsigned char*)d_in[36];
  const unsigned char* nodemask = (const unsigned char*)d_in[37];

  // ---- workspace carving
  char* wsb = (char*)d_ws;
  size_t off = 0;
  auto alloc = [&](size_t bytes) -> char* {
    char* p = wsb + off;
    off = (off + bytes + 255) & ~(size_t)255;
    return p;
  };
  const size_t SV = (size_t)NB * NN * NM * ND;  // 1M vector-channel elems
  const size_t SS = (size_t)NB * NN * ND;       // 32K scalar-channel elems
  const size_t SE = (size_t)NB * NN * NN * ND;  // 8.4M edge elems

  float*    leF   = (float*)alloc((size_t)NB * NM * ND * 4);
  float*    coordW= (float*)alloc(SV * 4);
  float*    vF    = (float*)alloc(SV * 4);
  _Float16* vH    = (_Float16*)alloc(SV * 2);
  _Float16* xH    = (_Float16*)alloc(SS * 2);
  float*    sbF   = (float*)alloc(SS * 4);
  _Float16* sbH   = (_Float16*)alloc(SS * 2);
  float*    s1F   = (float*)alloc(SS * 4);
  float*    s2F   = (float*)alloc(SS * 4);
  _Float16* v1T   = (_Float16*)alloc(SV * 2);
  _Float16* v2T   = (_Float16*)alloc(SV * 2);
  float*    elF   = (float*)alloc(SE * 4);
  _Float16* elT   = (_Float16*)alloc(SE * 2);
  float*    p1F   = (float*)alloc(SV * 4);
  float*    p2F   = (float*)alloc(SV * 4);
  float*    vpF   = (float*)alloc(SS * 4);
  _Float16* hT    = (_Float16*)alloc(SS * 2);
  float*    m1F   = (float*)alloc(SS * 4);
  float*    m2F   = (float*)alloc(SS * 4);
  _Float16* tsH   = (_Float16*)alloc(SS * 2);
  _Float16* tvH   = (_Float16*)alloc(SV * 2);  // (b,m,n,d) permuted
  float*    gF    = (float*)alloc(SS * 4);
  _Float16* wT    = (_Float16*)alloc(SV * 2);  // (b,d,m,j)
  _Float16* wgt   = (_Float16*)alloc((size_t)15 * NL * ND * ND * 2);
  float*    gsnF  = (float*)alloc(256);
  float*    XW    = (float*)alloc(SS * 4);
  float*    XpF   = (float*)alloc((size_t)NB * ND * 4);

  auto wslot = [&](int s) { return wgt + (size_t)s * NL * ND * ND; };
  auto gemm = [&](const _Float16* A, const _Float16* WT16, const float* bias,
                  float* oF, _Float16* oH, int mode, int relu, int rows, int rpb) {
    k_gemm64<<<rows / 32, 256, 0, stream>>>(A, WT16, bias, oF, oH, mode, relu, rpb);
  };

  // ---- prologue
  k_gsn<<<NB, NN, 0, stream>>>(nodemask, gsnF);
  k_lambda<<<NB, 256, 0, stream>>>(LambdaEmb, Wphi, bphi, Wrho, brho, LambdaMask, leF);
  k_coord<<<dim3(NN, NB), 256, 0, stream>>>(U, leF, coordW);
  k_copyf<<<(int)(SS / 256), 256, 0, stream>>>(X, XW, (int)SS);
  const float* wsrc[15] = {el_Ws1, el_Ws2, el_Wv1, el_Wv2, el_mW1, el_mW2,
                           mx_Wv1, mx_Wv2, mx_s1W1, mx_s1W2, mx_s2W1, mx_s2W2,
                           cv_W1, cv_W2, cv_Wv};
  for (int s = 0; s < 15; ++s)
    k_prepw<<<NL, 256, 0, stream>>>(wsrc[s], wslot(s));

  const int EL_LDS = 98304;  // eL + hL + s-tiles + 2 TDM-staged weight tiles

  // ---- layers
  for (int i = 0; i < NL; ++i) {
    const int wOff = i * ND * ND;
    k_vln<<<dim3(NN, NB), 256, 0, stream>>>(coordW, vF, vH);
    k_sln<<<dim3(NN, NB), ND, 0, stream>>>(XW, sbF, sbH);
    k_tohalf<<<(int)(SS / 256), 256, 0, stream>>>(XW, xH, (int)SS);
    // sv2el scalars + vectors
    gemm(xH, wslot(0) + wOff, el_bs1 + i * ND, s1F, nullptr, 0, 0, NB * NN, 0);
    gemm(xH, wslot(1) + wOff, el_bs2 + i * ND, s2F, nullptr, 0, 0, NB * NN, 0);
    gemm(vH, wslot(2) + wOff, nullptr, nullptr, v1T, 2, 0, NB * NN * NM, NN * NM);
    gemm(vH, wslot(3) + wOff, nullptr, nullptr, v2T, 2, 0, NB * NN * NM, NN * NM);
    k_el<<<dim3(16, 16, NB), 256, EL_LDS, stream>>>(
        v1T, v2T, s1F, s2F, wslot(4) + wOff, el_mb1 + i * ND, wslot(5) + wOff,
        el_mb2 + i * ND, gsnF, elF, elT, (i == 0) ? 1 : 0);
    // svMix
    gemm(vH, wslot(6) + wOff, nullptr, p1F, nullptr, 0, 0, NB * NN * NM, 0);
    gemm(vH, wslot(7) + wOff, nullptr, p2F, nullptr, 0, 0, NB * NN * NM, 0);
    k_vprod<<<dim3(NN, NB), ND, 0, stream>>>(p1F, p2F, vpF);
    gemm(sbH, wslot(8) + wOff, mx_s1b1 + i * ND, nullptr, hT, 1, 1, NB * NN, 0);
    gemm(hT, wslot(9) + wOff, mx_s1b2 + i * ND, m1F, nullptr, 0, 0, NB * NN, 0);
    gemm(sbH, wslot(10) + wOff, mx_s2b1 + i * ND, nullptr, hT, 1, 1, NB * NN, 0);
    gemm(hT, wslot(11) + wOff, mx_s2b2 + i * ND, m2F, nullptr, 0, 0, NB * NN, 0);
    k_ts<<<(int)(SS / 256), 256, 0, stream>>>(sbF, m1F, vpF, tsH, (int)SS);
    k_tv<<<(int)(SV / 256), 256, 0, stream>>>(vF, m2F, tvH, (int)SV);
    // DirCFConv
    gemm(tsH, wslot(12) + wOff, cv_b1 + i * ND, nullptr, hT, 1, 1, NB * NN, 0);
    gemm(hT, wslot(13) + wOff, cv_b2 + i * ND, gF, nullptr, 0, 0, NB * NN, 0);
    gemm(tvH, wslot(14) + wOff, nullptr, nullptr, wT, 2, 0, NB * NN * NM, NN * NM);
    k_vo<<<NB * ND, 256, 0, stream>>>(elT, wT, coordW);
    k_so<<<dim3(NN, NB), 256, 0, stream>>>(elF, gF, XW);
  }

  // ---- head
  k_pool<<<NB, ND, 0, stream>>>(XW, nodemask, XpF);
  k_pred<<<NB, ND, 0, stream>>>(XpF, pred_W1, pred_b1, pred_W2, pred_b2,
                                (float*)d_out);
}
